// MultiheadSelfAttention_57432302682375
// MI455X (gfx1250) — compile-verified, matrix operand order checked by reading, not run on hardware
//
#include <hip/hip_runtime.h>

#define DEV __device__ __forceinline__

typedef __attribute__((ext_vector_type(16))) _Float16 v16h;
typedef __attribute__((ext_vector_type(8)))  float    v8f;
typedef __attribute__((ext_vector_type(4)))  unsigned int u32x4;
typedef __attribute__((ext_vector_type(8)))  int      i32x8;
typedef __attribute__((ext_vector_type(4)))  int      i32x4;

static constexpr int SEQ = 4096;
static constexpr int DM  = 1024;
static constexpr int NH  = 16;
static constexpr int DK  = 64;   // head dim

#if __has_builtin(__builtin_amdgcn_tensor_load_to_lds) && __has_builtin(__builtin_amdgcn_s_wait_tensorcnt)
#define USE_TDM 1
#else
#define USE_TDM 0
#endif

union HFrag { v16h v; uint4 u[2]; _Float16 h[16]; };

// Gather a 16x32 f16 operand fragment from an LDS tile laid out [row][32 cols]
// (row stride in halves). Matches the CDNA5 16-bit A/B layout:
// lane<16  -> row=lane,    K chunks {0..7, 16..23}
// lane>=16 -> row=lane-16, K chunks {8..15, 24..31}
DEV v16h ldsFrag(const _Float16* base, int stride, int rowBase, int lane) {
    int r    = rowBase + (lane & 15);
    int koff = (lane >> 4) << 3;              // 0 or 8
    const _Float16* p = base + r * stride + koff;
    HFrag f;
    f.u[0] = *(const uint4*)(p);              // K = koff .. koff+7
    f.u[1] = *(const uint4*)(p + 16);         // K = koff+16 .. koff+23
    return f.v;
}

DEV v8f wmma16(v16h a, v16h b, v8f c) {
    return __builtin_amdgcn_wmma_f32_16x16x32_f16(
        /*neg_a=*/false, a, /*neg_b=*/false, b,
        /*c_mod=*/(short)0, c, /*reuse_a=*/false, /*reuse_b=*/false);
}

#if USE_TDM
// Tensor Data Mover: DMA a 2D f16 tile (tile_w halves x tile_h rows, global
// row stride in halves) into LDS, inserting pad_amount DWORD padding every
// pad_interval DWORDs (D# group1 pad fields) to produce padded LDS rows.
// Issue from ONE wave only; EXEC is ignored by TDM; tracked by TENSORcnt.
DEV void tdm_load_2d(const _Float16* gsrc, _Float16* lds_dst,
                     int tile_w, int tile_h, int row_stride,
                     int pad_interval_code, int pad_amount_code) {
    unsigned long long ga = (unsigned long long)(uintptr_t)gsrc;
    unsigned int       la = (unsigned int)(uintptr_t)lds_dst;  // LDS byte offset

    u32x4 g0;
    g0[0] = 1u;                                         // count=1, user D#
    g0[1] = la;                                         // lds_addr
    g0[2] = (unsigned int)ga;                           // global_addr[31:0]
    g0[3] = (unsigned int)((ga >> 32) & 0x1FFFFFFu)     // global_addr[56:32]
          | (2u << 30);                                 // type = 2 ("image")

    i32x8 g1;
    g1[0] = (1 << 16)                                   // data_size = 2 bytes
          | (1 << 20)                                   // pad_enable
          | (pad_interval_code << 22)
          | (pad_amount_code << 25);
    g1[1] = tile_w << 16;                               // tensor_dim0[15:0]
    g1[2] = tile_h << 16;                               // tensor_dim1[15:0]
    g1[3] = tile_w << 16;                               // tile_dim0
    g1[4] = tile_h;                                     // tile_dim1 (tile_dim2=0)
    g1[5] = row_stride;                                 // tensor_dim0_stride[31:0]
    g1[6] = 0;
    g1[7] = 0;

    i32x4 z4 = {};
#if defined(__clang_major__) && (__clang_major__ >= 23)
    i32x8 z8 = {};
    __builtin_amdgcn_tensor_load_to_lds(g0, g1, z4, z4, z8, 0);
#else
    __builtin_amdgcn_tensor_load_to_lds(g0, g1, z4, z4, 0);
#endif
}
#endif

// ---------------------------------------------------------------- f32 -> f16
__global__ void k_cvt(const float* __restrict__ in, _Float16* __restrict__ out, int n4) {
    int i = blockIdx.x * blockDim.x + threadIdx.x;
    if (i >= n4) return;
    float4 v = ((const float4*)in)[i];
    union { _Float16 h[4]; uint2 u; } o;
    o.h[0] = (_Float16)v.x; o.h[1] = (_Float16)v.y;
    o.h[2] = (_Float16)v.z; o.h[3] = (_Float16)v.w;
    ((uint2*)out)[i] = o.u;
}

// ------------------------------------------------- C = A(MxK) * B(NxK)^T
// 128x256 tile per 256-thread block; 8 waves of 64x64 (16 WMMAs per k-slab);
// TDM double-buffered LDS staging: next slab DMA overlaps current WMMAs.
template<bool F32OUT>
__global__ void k_gemm_nt(const _Float16* __restrict__ A,
                          const _Float16* __restrict__ B,
                          void* __restrict__ C, int M, int N, int K) {
    __shared__ _Float16 As[2][128 * 40];   // 32-wide k-slab + 8-half pad
    __shared__ _Float16 Bs[2][256 * 40];

    const int t    = threadIdx.x;
    const int lane = t & 31;
    const int w    = t >> 5;
    const int wm   = w >> 2;               // 2 waves along M
    const int wn   = w & 3;                // 4 waves along N
    const int m0   = blockIdx.y * 128;
    const int n0   = blockIdx.x * 256;

    v8f acc[4][4] = {};

    auto stage = [&](int buf, int k0) {
#if USE_TDM
        if (t < 32) {   // wave 0 drives the Tensor Data Mover
            tdm_load_2d(A + (size_t)m0 * K + k0, &As[buf][0], 32, 128, K, 3, 3);
            tdm_load_2d(B + (size_t)n0 * K + k0, &Bs[buf][0], 32, 256, K, 3, 3);
        }
#else
        {   // manual fallback staging
            int rowA = t >> 1, c16 = (t & 1) << 4;
            const _Float16* gA = A + (size_t)(m0 + rowA) * K + k0 + c16;
            *(uint4*)&As[buf][rowA * 40 + c16]     = *(const uint4*)gA;
            *(uint4*)&As[buf][rowA * 40 + c16 + 8] = *(const uint4*)(gA + 8);
            const _Float16* gB = B + (size_t)(n0 + t) * K + k0;
            for (int j = 0; j < 4; ++j)
                *(uint4*)&Bs[buf][t * 40 + j * 8] = *(const uint4*)(gB + j * 8);
        }
#endif
    };

    stage(0, 0);
#if USE_TDM
    if (t < 32) __builtin_amdgcn_s_wait_tensorcnt(0);
#endif
    __syncthreads();

    int cur = 0;
    for (int k0 = 0; k0 < K; k0 += 32) {
        const int nxt = cur ^ 1;
        if (k0 + 32 < K) stage(nxt, k0 + 32);   // DMA next slab during compute

        v16h af[4], bf[4];
        for (int i = 0; i < 4; ++i) af[i] = ldsFrag(&As[cur][0], 40, wm * 64 + i * 16, lane);
        for (int j = 0; j < 4; ++j) bf[j] = ldsFrag(&Bs[cur][0], 40, wn * 64 + j * 16, lane);
        for (int i = 0; i < 4; ++i)
            for (int j = 0; j < 4; ++j)
                acc[i][j] = wmma16(af[i], bf[j], acc[i][j]);

#if USE_TDM
        if (t < 32 && k0 + 32 < K) __builtin_amdgcn_s_wait_tensorcnt(0);
#endif
        __syncthreads();
        cur = nxt;
    }

    // C/D layout: VGPR r, lanes 0-15 -> M=r, lanes 16-31 -> M=r+8, N=lane&15
    const int hi = lane >> 4, ln = lane & 15;
    for (int i = 0; i < 4; ++i)
        for (int j = 0; j < 4; ++j)
            for (int r = 0; r < 8; ++r) {
                int row = m0 + wm * 64 + i * 16 + r + hi * 8;
                int col = n0 + wn * 64 + j * 16 + ln;
                float v = acc[i][j][r];
                if constexpr (F32OUT) ((float*)C)[(size_t)row * N + col] = v;
                else ((_Float16*)C)[(size_t)row * N + col] = (_Float16)v;
            }
}

// ---------------------------------------------------------------- RoPE (Q,K)
__global__ void k_rope(_Float16* __restrict__ Q, _Float16* __restrict__ Kc,
                       const float* __restrict__ cosT, const float* __restrict__ sinT,
                       const int* __restrict__ pos) {
    int i  = blockIdx.x * blockDim.x + threadIdx.x;   // SEQ * (DM/2) threads
    int s  = i >> 9;
    int pj = i & 511;
    int h  = pj >> 5, j = pj & 31;                    // 32 pairs per head
    int p  = pos[s];
    float c  = cosT[p * 32 + j];
    float sn = sinT[p * 32 + j];
    size_t idx = (size_t)s * DM + h * DK + 2 * j;
    float q1 = (float)Q[idx], q2 = (float)Q[idx + 1];
    Q[idx]     = (_Float16)(q1 * c - q2 * sn);
    Q[idx + 1] = (_Float16)(q1 * sn + q2 * c);
    float k1 = (float)Kc[idx], k2 = (float)Kc[idx + 1];
    Kc[idx]     = (_Float16)(k1 * c - k2 * sn);
    Kc[idx + 1] = (_Float16)(k1 * sn + k2 * c);
}

// ------------------------------------------------ causal flash attention
// grid = (SEQ/128, NH); 256 threads; wave w owns query rows [w*16, w*16+16).
// Double-buffered K/V staging: next tile's TDM DMA + transposed-V stores
// overlap the current tile's WMMA + softmax.
__global__ void k_attn(const _Float16* __restrict__ Q, const _Float16* __restrict__ K,
                       const _Float16* __restrict__ V, _Float16* __restrict__ O) {
    __shared__ _Float16 Qs[128 * 72];        // 128 x 64 (+8 pad)
    __shared__ _Float16 Ks[2][32 * 72];      // 32 keys x 64, double buffered
    __shared__ _Float16 VTs[2][64 * 40];     // V^T: 64 dims x 32 keys (+8 pad)
    __shared__ _Float16 Ps[8 * 16 * 40];     // per-wave 16 x 32 P strip

    const int t = threadIdx.x, lane = t & 31, w = t >> 5;
    const int head = blockIdx.y, qt = blockIdx.x;
    const int q0 = qt * 128;
    const size_t hoff = (size_t)head * DK;

    auto stageKV = [&](int buf, int kt) {
#if USE_TDM
        if (t < 32)
            tdm_load_2d(K + (size_t)(kt * 32) * DM + hoff, &Ks[buf][0], 64, 32, DM, 4, 3);
#else
        {   // stage K tile row-major
            int key = t >> 3, d0 = (t & 7) * 8;
            const _Float16* gk = K + (size_t)(kt * 32 + key) * DM + hoff + d0;
            *(uint4*)&Ks[buf][key * 72 + d0] = *(const uint4*)gk;
        }
#endif
        {   // stage V tile transposed (TDM cannot transpose)
            int key = t >> 3, d0 = (t & 7) * 8;
            const _Float16* gv = V + (size_t)(kt * 32 + key) * DM + hoff + d0;
            HFrag tmp; tmp.u[0] = *(const uint4*)gv;
            for (int j = 0; j < 8; ++j) VTs[buf][(d0 + j) * 40 + key] = tmp.h[j];
        }
    };

#if USE_TDM
    if (t < 32)
        tdm_load_2d(Q + (size_t)q0 * DM + hoff, Qs, 64, 128, DM, 4, 3);
#else
    {   // stage the 128x64 Q tile manually
        int row = t >> 1, d0 = (t & 1) * 32;
        const _Float16* g = Q + (size_t)(q0 + row) * DM + hoff + d0;
        for (int j = 0; j < 4; ++j)
            *(uint4*)&Qs[row * 72 + d0 + j * 8] = *(const uint4*)(g + j * 8);
    }
#endif
    stageKV(0, 0);
#if USE_TDM
    if (t < 32) __builtin_amdgcn_s_wait_tensorcnt(0);   // Q + first K DMAs
#endif
    __syncthreads();

    const int hi = lane >> 4, ln = lane & 15;
    const int qrb = w * 16;

    float mrow[8], lrow[8];
    v8f o[4] = {};
    for (int r = 0; r < 8; ++r) { mrow[r] = -__builtin_inff(); lrow[r] = 0.f; }

    const int ktiles = (qt + 1) * 4;      // keys [0, q0+128), 32 per tile
    int cur = 0;
    for (int kt = 0; kt < ktiles; ++kt) {
        const int nxt = cur ^ 1;
        if (kt + 1 < ktiles) stageKV(nxt, kt + 1);   // overlap DMA with compute

        // S (16 x 32) = Q_strip (16x64) * K_tile^T ; d split into two k-steps
        v8f s0 = {}, s1 = {};
        for (int step = 0; step < 2; ++step) {
            v16h a  = ldsFrag(Qs + step * 32, 72, qrb, lane);
            v16h b0 = ldsFrag(&Ks[cur][0] + step * 32, 72, 0,  lane);
            v16h b1 = ldsFrag(&Ks[cur][0] + step * 32, 72, 16, lane);
            s0 = wmma16(a, b0, s0);
            s1 = wmma16(a, b1, s1);
        }

        // online softmax (row stats replicated across the owning 16-lane half)
        float p0[8], p1[8];
        for (int r = 0; r < 8; ++r) {
            int rowg = q0 + qrb + r + hi * 8;
            int c0 = kt * 32 + ln, c1 = kt * 32 + 16 + ln;
            float v0 = (c0 <= rowg) ? s0[r] * 0.125f : -__builtin_inff();
            float v1 = (c1 <= rowg) ? s1[r] * 0.125f : -__builtin_inff();
            float tmax = fmaxf(v0, v1);
            for (int d = 1; d < 16; d <<= 1) tmax = fmaxf(tmax, __shfl_xor(tmax, d, 32));
            float mnew = fmaxf(mrow[r], tmax);
            float f  = __expf(mrow[r] - mnew);
            float e0 = __expf(v0 - mnew);
            float e1 = __expf(v1 - mnew);
            float rs = e0 + e1;
            for (int d = 1; d < 16; d <<= 1) rs += __shfl_xor(rs, d, 32);
            lrow[r] = lrow[r] * f + rs;
            mrow[r] = mnew;
            for (int tn = 0; tn < 4; ++tn) o[tn][r] *= f;
            p0[r] = e0; p1[r] = e1;
        }

        // round-trip P through LDS: C-layout -> A-fragment layout
        _Float16* Pw = Ps + w * (16 * 40);
        for (int r = 0; r < 8; ++r) {
            Pw[(r + hi * 8) * 40 + ln]      = (_Float16)p0[r];
            Pw[(r + hi * 8) * 40 + 16 + ln] = (_Float16)p1[r];
        }

        // O (16x64) += P (16x32) * V (32x64) : one k-step, 4 n-tiles
        v16h ap = ldsFrag(Pw, 40, 0, lane);
        for (int tn = 0; tn < 4; ++tn) {
            v16h bv = ldsFrag(&VTs[cur][0], 40, tn * 16, lane);
            o[tn] = wmma16(ap, bv, o[tn]);
        }

#if USE_TDM
        if (t < 32 && kt + 1 < ktiles) __builtin_amdgcn_s_wait_tensorcnt(0);
#endif
        __syncthreads();
        cur = nxt;
    }

    for (int r = 0; r < 8; ++r) {
        float inv = 1.f / lrow[r];
        int rowg = q0 + qrb + r + hi * 8;
        for (int tn = 0; tn < 4; ++tn)
            O[(size_t)rowg * DM + hoff + tn * 16 + ln] = (_Float16)(o[tn][r] * inv);
    }
}

// ---------------------------------------------------------------- launcher
extern "C" void kernel_launch(void* const* d_in, const int* in_sizes, int n_in,
                              void* d_out, int out_size, void* d_ws, size_t ws_size,
                              hipStream_t stream) {
    const float* x    = (const float*)d_in[0];
    const float* Wq   = (const float*)d_in[1];
    const float* Wk   = (const float*)d_in[2];
    const float* Wv   = (const float*)d_in[3];
    const float* Wo   = (const float*)d_in[4];
    const float* cosT = (const float*)d_in[5];
    const float* sinT = (const float*)d_in[6];
    const int*   pos  = (const int*)d_in[7];

    _Float16* ws = (_Float16*)d_ws;
    size_t off = 0;
    _Float16* xb  = ws + off; off += (size_t)SEQ * DM;
    _Float16* wqb = ws + off; off += (size_t)DM * DM;
    _Float16* wkb = ws + off; off += (size_t)DM * DM;
    _Float16* wvb = ws + off; off += (size_t)DM * DM;
    _Float16* wob = ws + off; off += (size_t)DM * DM;
    _Float16* Qh  = ws + off; off += (size_t)SEQ * DM;
    _Float16* Kh  = ws + off; off += (size_t)SEQ * DM;
    _Float16* Vh  = ws + off; off += (size_t)SEQ * DM;
    _Float16* Ah  = ws + off; off += (size_t)SEQ * DM;

    dim3 blk(256);
    k_cvt<<<(SEQ * DM / 4 + 255) / 256, blk, 0, stream>>>(x,  xb,  SEQ * DM / 4);
    k_cvt<<<(DM * DM / 4 + 255) / 256, blk, 0, stream>>>(Wq, wqb, DM * DM / 4);
    k_cvt<<<(DM * DM / 4 + 255) / 256, blk, 0, stream>>>(Wk, wkb, DM * DM / 4);
    k_cvt<<<(DM * DM / 4 + 255) / 256, blk, 0, stream>>>(Wv, wvb, DM * DM / 4);
    k_cvt<<<(DM * DM / 4 + 255) / 256, blk, 0, stream>>>(Wo, wob, DM * DM / 4);

    dim3 gg(DM / 256, SEQ / 128);
    k_gemm_nt<false><<<gg, blk, 0, stream>>>(xb, wqb, Qh, SEQ, DM, DM);
    k_gemm_nt<false><<<gg, blk, 0, stream>>>(xb, wkb, Kh, SEQ, DM, DM);
    k_gemm_nt<false><<<gg, blk, 0, stream>>>(xb, wvb, Vh, SEQ, DM, DM);

    k_rope<<<(SEQ * (DM / 2)) / 256, blk, 0, stream>>>(Qh, Kh, cosT, sinT, pos);

    dim3 ga(SEQ / 128, NH);
    k_attn<<<ga, blk, 0, stream>>>(Qh, Kh, Vh, Ah);

    k_gemm_nt<true><<<gg, blk, 0, stream>>>(Ah, wob, d_out, SEQ, DM, DM);
}
